// Block_CD_18416819766013
// MI455X (gfx1250) — matrix operand order repaired and checked by reading, not compiled, not run on hardware
//
#include <hip/hip_runtime.h>

typedef __bf16 bf16;
typedef __attribute__((ext_vector_type(16))) __bf16 v16bf;
typedef __attribute__((ext_vector_type(8)))  float  v8f;

#define L     25
#define ES    257            // xe/ye row stride (floats)  -> bank-rotating
#define AS    264            // a (bf16) row stride        -> 16B aligned, bank-friendly
#define QS    776            // qkv (bf16) row stride
#define ATT_SCALE 0.1767766952966369f  // 1/sqrt(32)

// bf16 weight workspace layout (elements)
#define OFF_QX 0
#define OFF_PX 196608
#define OFF_QY 262144
#define OFF_PY 458752
#define W_TOTAL 524288

struct SMem {
    float img[2][3][7][7];     // zero-padded input images
    float xe[L * ES];          // conv-x embed, later (+ proj result) = output chans 0..255
    float ye[L * ES];          // conv-y embed, later output chans 256..511
    bf16  a[32 * AS];          // LN output / attention output (A matrix, rows 25..31 zero)
    bf16  qkv[32 * QS];        // q|k|v, [token][s*256 + h*32 + d]
};

__global__ void convert_weights(const float* qx, const float* px,
                                const float* qy, const float* py, bf16* ws) {
    int i = blockIdx.x * 256 + threadIdx.x;
    if (i < OFF_PX)            ws[i] = (bf16)qx[i];
    else if (i < OFF_QY)       ws[i] = (bf16)px[i - OFF_PX];
    else if (i < OFF_PY)       ws[i] = (bf16)qy[i - OFF_QY];
    else if (i < W_TOTAL)      ws[i] = (bf16)py[i - OFF_PY];
}

__global__ __launch_bounds__(256) void fused_block(
        const float* __restrict__ x, const float* __restrict__ y,
        const float* __restrict__ c1w, const float* __restrict__ c1b,
        const float* __restrict__ c2w, const float* __restrict__ c2b,
        const float* __restrict__ pos,
        const float* __restrict__ lng, const float* __restrict__ lnb,
        const float* __restrict__ pbx, const float* __restrict__ pby,
        const bf16*  __restrict__ wbf, float* __restrict__ out) {
    __shared__ SMem sm;
    const int tid  = threadIdx.x;
    const int lane = tid & 31;
    const int wave = tid >> 5;
    const int hi   = lane >> 4;   // 0/1: which K-half this lane holds
    const int lo   = lane & 15;
    const int b    = blockIdx.x;

    // ---- stage images into zero-padded LDS ----
    float* imgf = &sm.img[0][0][0][0];
    for (int i = tid; i < 294; i += 256) imgf[i] = 0.f;
    __syncthreads();
    for (int i = tid; i < 150; i += 256) {
        int s = i / 75, r = i % 75, ci = r / 25, p = r % 25;
        const float* src = s ? y : x;
        sm.img[s][ci][p / 5 + 1][p % 5 + 1] = src[(size_t)b * 75 + ci * 25 + p];
    }
    __syncthreads();

    // ---- 3x3 conv, thread = output channel ----
    {
        const int c = tid;
        for (int s = 0; s < 2; ++s) {
            const float* cw = s ? c2w : c1w;
            float bias = (s ? c2b : c1b)[c];
            float* e = s ? sm.ye : sm.xe;
            float w[27];
            #pragma unroll
            for (int k = 0; k < 27; ++k) w[k] = cw[c * 27 + k];
            for (int py = 0; py < 5; ++py)
                for (int px = 0; px < 5; ++px) {
                    float acc = bias;
                    #pragma unroll
                    for (int ci = 0; ci < 3; ++ci)
                        #pragma unroll
                        for (int kh = 0; kh < 3; ++kh)
                            #pragma unroll
                            for (int kw = 0; kw < 3; ++kw)
                                acc += sm.img[s][ci][py + kh][px + kw] * w[ci * 9 + kh * 3 + kw];
                    e[(py * 5 + px) * ES + c] = acc;
                }
        }
    }
    __syncthreads();

    // ---- per-stream: LN -> qkv GEMM -> attention -> proj GEMM (residual in LDS) ----
    for (int s = 0; s < 2; ++s) {
        float* e = s ? sm.ye : sm.xe;
        const bf16* qw = wbf + (s ? OFF_QY : OFF_QX);
        const bf16* pw = wbf + (s ? OFF_PY : OFF_PX);
        const float* pb = s ? pby : pbx;

        // LayerNorm(x + pos_embed) -> bf16 A-buffer; wave per token
        for (int p = wave; p < L; p += 8) {
            float t[8], sum = 0.f, sq = 0.f;
            const int c0 = lane * 8;
            #pragma unroll
            for (int i = 0; i < 8; ++i) {
                float v = e[p * ES + c0 + i] + pos[p * 256 + c0 + i];
                t[i] = v; sum += v; sq += v * v;
            }
            #pragma unroll
            for (int off = 16; off >= 1; off >>= 1) {
                sum += __shfl_xor(sum, off, 32);
                sq  += __shfl_xor(sq,  off, 32);
            }
            float mu = sum * (1.f / 256.f);
            float rs = rsqrtf(sq * (1.f / 256.f) - mu * mu + 1e-5f);
            #pragma unroll
            for (int i = 0; i < 8; ++i)
                sm.a[p * AS + c0 + i] = (bf16)((t[i] - mu) * rs * lng[c0 + i] + lnb[c0 + i]);
        }
        for (int i = tid; i < 7 * AS; i += 256) sm.a[L * AS + i] = (bf16)0.f;  // pad rows
        __syncthreads();

        // qkv GEMM: [32x256] x [768x256]^T via v_wmma_f32_16x16x32_bf16
        for (int t = 0; t < 12; ++t) {
            const int nt = wave + (t >> 1) * 8;   // N tile 0..47
            const int mt = t & 1;                 // M tile 0..1
            v8f acc = {0.f, 0.f, 0.f, 0.f, 0.f, 0.f, 0.f, 0.f};
            const bf16* brow = qw   + (nt * 16 + lo) * 256;
            const bf16* arow = sm.a + (mt * 16 + lo) * AS;
            #pragma unroll
            for (int ks = 0; ks < 8; ++ks) {
                union { v16bf v; float4 f[2]; } A, Bm;
                const bf16* ap = arow + ks * 32 + hi * 16;
                A.f[0] = *(const float4*)(ap);
                A.f[1] = *(const float4*)(ap + 8);
                const bf16* bp = brow + ks * 32 + hi * 16;
                Bm.f[0] = *(const float4*)(bp);
                Bm.f[1] = *(const float4*)(bp + 8);
                acc = __builtin_amdgcn_wmma_f32_16x16x32_bf16(
                          false, A.v, false, Bm.v, (short)0, acc, false, false);
            }
            #pragma unroll
            for (int r = 0; r < 8; ++r)
                sm.qkv[(mt * 16 + hi * 8 + r) * QS + nt * 16 + lo] = (bf16)acc[r];
        }
        __syncthreads();

        // attention: wave = head; lanes = tokens for scores, = dims for output
        {
            const int h = wave;
            union { float4 f[4]; bf16 v[32]; } K4, Q4;
            float kreg[32], vt[L];
            const bf16* kp = sm.qkv + lane * QS + 256 + h * 32;
            K4.f[0] = *(const float4*)(kp);      K4.f[1] = *(const float4*)(kp + 8);
            K4.f[2] = *(const float4*)(kp + 16); K4.f[3] = *(const float4*)(kp + 24);
            #pragma unroll
            for (int d = 0; d < 32; ++d) kreg[d] = (float)K4.v[d];
            #pragma unroll
            for (int j = 0; j < L; ++j)
                vt[j] = (float)sm.qkv[j * QS + 512 + h * 32 + lane];

            for (int i = 0; i < L; ++i) {
                const bf16* qp = sm.qkv + i * QS + h * 32;
                Q4.f[0] = *(const float4*)(qp);      Q4.f[1] = *(const float4*)(qp + 8);
                Q4.f[2] = *(const float4*)(qp + 16); Q4.f[3] = *(const float4*)(qp + 24);
                float sc = 0.f;
                #pragma unroll
                for (int d = 0; d < 32; ++d) sc += (float)Q4.v[d] * kreg[d];
                sc = (lane < L) ? sc * ATT_SCALE : -1e30f;
                float m = sc;
                #pragma unroll
                for (int off = 16; off >= 1; off >>= 1) m = fmaxf(m, __shfl_xor(m, off, 32));
                float pexp = (lane < L) ? __expf(sc - m) : 0.f;
                float ssum = pexp;
                #pragma unroll
                for (int off = 16; off >= 1; off >>= 1) ssum += __shfl_xor(ssum, off, 32);
                float pn = pexp / ssum;
                float o = 0.f;
                #pragma unroll
                for (int j = 0; j < L; ++j) o += __shfl(pn, j, 32) * vt[j];
                sm.a[i * AS + h * 32 + lane] = (bf16)o;  // rows 25..31 stay zero
            }
        }
        __syncthreads();

        // proj GEMM [32x256]x[256x256]^T, accumulate (+bias) into conv-embed residual
        for (int t = 0; t < 4; ++t) {
            const int nt = wave + (t >> 1) * 8;
            const int mt = t & 1;
            v8f acc = {0.f, 0.f, 0.f, 0.f, 0.f, 0.f, 0.f, 0.f};
            const bf16* brow = pw   + (nt * 16 + lo) * 256;
            const bf16* arow = sm.a + (mt * 16 + lo) * AS;
            #pragma unroll
            for (int ks = 0; ks < 8; ++ks) {
                union { v16bf v; float4 f[2]; } A, Bm;
                const bf16* ap = arow + ks * 32 + hi * 16;
                A.f[0] = *(const float4*)(ap);
                A.f[1] = *(const float4*)(ap + 8);
                const bf16* bp = brow + ks * 32 + hi * 16;
                Bm.f[0] = *(const float4*)(bp);
                Bm.f[1] = *(const float4*)(bp + 8);
                acc = __builtin_amdgcn_wmma_f32_16x16x32_bf16(
                          false, A.v, false, Bm.v, (short)0, acc, false, false);
            }
            const int col = nt * 16 + lo;
            const float bias = pb[col];
            #pragma unroll
            for (int r = 0; r < 8; ++r) {
                const int row = mt * 16 + hi * 8 + r;
                if (row < L) e[row * ES + col] += acc[r] + bias;
            }
        }
        __syncthreads();
    }

    // ---- coalesced channel-major store: out[b][ch][p], ch<256 from xe else ye ----
    {
        const size_t base = (size_t)b * 12800;
        for (int i = tid; i < 12800; i += 256) {
            const int ch = i / 25, p = i - ch * 25;
            out[base + i] = (ch < 256) ? sm.xe[p * ES + ch] : sm.ye[p * ES + (ch - 256)];
        }
    }
}

extern "C" void kernel_launch(void* const* d_in, const int* in_sizes, int n_in,
                              void* d_out, int out_size, void* d_ws, size_t ws_size,
                              hipStream_t stream) {
    (void)in_sizes; (void)n_in; (void)out_size; (void)ws_size;
    const float* x   = (const float*)d_in[0];
    const float* y   = (const float*)d_in[1];
    const float* c1w = (const float*)d_in[2];
    const float* c1b = (const float*)d_in[3];
    const float* c2w = (const float*)d_in[4];
    const float* c2b = (const float*)d_in[5];
    const float* pos = (const float*)d_in[6];
    const float* lng = (const float*)d_in[7];
    const float* lnb = (const float*)d_in[8];
    const float* qxw = (const float*)d_in[9];
    const float* pxw = (const float*)d_in[10];
    const float* pxb = (const float*)d_in[11];
    const float* qyw = (const float*)d_in[12];
    const float* pyw = (const float*)d_in[13];
    const float* pyb = (const float*)d_in[14];
    bf16* wbf = (bf16*)d_ws;

    convert_weights<<<W_TOTAL / 256, 256, 0, stream>>>(qxw, pxw, qyw, pyw, wbf);
    fused_block<<<8192, 256, 0, stream>>>(x, y, c1w, c1b, c2w, c2b, pos,
                                          lng, lnb, pxb, pyb, wbf, (float*)d_out);
}
